// SSM_54193897341495
// MI455X (gfx1250) — matrix operand order repaired and checked by reading, not compiled, not run on hardware
//
#include <hip/hip_runtime.h>
#include <hip/hip_bf16.h>
#include <utility>

// ---------------------------------------------------------------------------
// Hierarchical RSSM forward loss on gfx1250 (MI455X).
// All GEMM-shaped math (convs as implicit GEMM, deconvs as gather-GEMM, FCs,
// Gaussian heads) runs through one WMMA kernel (v_wmma_f32_16x16x32_f16).
//  - Weights pre-packed into the WMMA B-fragment layout [kb][n][k%32],
//    N/K zero-padded to 32: B fragments are guard-free global_load_b128.
//  - A tiles gathered branchlessly (clamped address + cndmask) into a
//    double-buffered LDS tile; fragments read as ds_load_b128.  Plain-matrix
//    mode uses vectorized b128 loads on interior K-blocks.
//  - Block tile 64(M) x 32(N), K-step 32, 4 waves, 2 WMMA per wave per step.
//  - Activation templated; bias hoisted; next-K B block prefetched.
// ---------------------------------------------------------------------------

typedef __attribute__((ext_vector_type(16))) _Float16 v16h;
typedef __attribute__((ext_vector_type(8)))  _Float16 v8h;
typedef __attribute__((ext_vector_type(8)))  float    v8f;
typedef __attribute__((ext_vector_type(4)))  float    v4f;

#define ACT_NONE     0
#define ACT_RELU     1
#define ACT_SOFTPLUS 2

struct GemmP {
  const float*    A;     // activation source (f32)
  const _Float16* W;     // packed weights, swizzled [kb][n][k%32], padded
  const float*    bias;  // [N], always valid
  float*          C;     // output [M, N] f32 (NHWC for convs)
  int M, N, K, Npad;
  // conv/deconv gather params (KH=KW=4, stride=2)
  int IC, iclog, IH, IW, OHW;
  int in_nchw;           // first conv reads NCHW input x
};

// AMODE: 0 = plain f32 row-major A, 1 = conv gather, 2 = deconv gather
template <int AMODE, int ACT>
__global__ void __launch_bounds__(128) gemm_wmma_k(GemmP p) {
  __shared__ alignas(16) _Float16 As[2][64 * 32];

  const int t    = threadIdx.x;
  const int wave = t >> 5;
  const int lane = t & 31;
  const int lmod = lane & 15;
  const int lhi  = lane >> 4;

  const int tilesN = p.Npad >> 5;
  const int mt = blockIdx.x / tilesN;
  const int nt = blockIdx.x % tilesN;
  const int m0 = mt << 6;          // 64-row block tile
  const int n0 = nt << 5;          // 32-col block tile

  // ---- A staging coords (constant over K loop) ----
  const int arowLocal = t & 63;            // local row 0..63
  const int akbase    = (t >> 6) << 4;     // 0 or 16
  const int arow  = m0 + arowLocal;
  const bool mval = arow < p.M;
  const int mc    = min(arow, p.M - 1);
  int gb = 0, goh = 0, gow = 0;
  if (AMODE != 0) {
    const int ohw = p.OHW;
    gb  = mc / (ohw * ohw);
    int r = mc % (ohw * ohw);
    goh = r / ohw;
    gow = r % ohw;
  }
  const bool k4ok = (p.K & 3) == 0;        // 16B alignment of row chunks

  v8f acc0 = {}, acc1 = {};
  const int nKb = (p.K + 31) >> 5;
  const unsigned wstep = (unsigned)p.Npad * 32u;

  for (int ib = 0; ib < nKb; ++ib) {
    const int kb = ib << 5;
    const int pp = ib & 1;

    // -------- stage A tile --------
    v8h alo, ahi;
    const int kstart = kb + akbase;
    if (AMODE == 0 && mval && k4ok && (kstart + 16 <= p.K)) {
      // interior fast path: 4x global_load_b128, then convert
      const v4f* src = (const v4f*)(p.A + (unsigned)mc * (unsigned)p.K + kstart);
      float fv[16];
      *(v4f*)&fv[0]  = src[0];
      *(v4f*)&fv[4]  = src[1];
      *(v4f*)&fv[8]  = src[2];
      *(v4f*)&fv[12] = src[3];
#pragma unroll
      for (int j = 0; j < 16; ++j) {
        const _Float16 hv = (_Float16)fv[j];
        if (j < 8) alo[j] = hv; else ahi[j - 8] = hv;
      }
    } else {
      // branchless clamped gather (f32 -> f16)
#pragma unroll
      for (int j = 0; j < 16; ++j) {
        const int k  = kstart + j;
        const int kc = min(k, p.K - 1);
        bool valid = mval && (k < p.K);
        unsigned idx;
        if (AMODE == 0) {
          idx = (unsigned)mc * (unsigned)p.K + (unsigned)kc;
        } else {
          int ic, q;
          if (p.iclog >= 0) { ic = kc & (p.IC - 1); q = kc >> p.iclog; }
          else              { q = kc / p.IC; ic = kc - q * p.IC; }
          const int kh = q >> 2, kw = q & 3;
          int ih, iw;
          if (AMODE == 1) {                    // stride-2 SAME conv, pad 1
            ih = goh * 2 + kh - 1;
            iw = gow * 2 + kw - 1;
          } else {                             // stride-2 transposed conv
            const int th = goh + 1 - kh;
            const int tw = gow + 1 - kw;
            valid = valid && (th >= 0) && (tw >= 0) && !(th & 1) && !(tw & 1);
            ih = th >> 1;
            iw = tw >> 1;
          }
          valid = valid && ((unsigned)ih < (unsigned)p.IH)
                        && ((unsigned)iw < (unsigned)p.IW);
          const int ihc = min(max(ih, 0), p.IH - 1);
          const int iwc = min(max(iw, 0), p.IW - 1);
          if (AMODE == 1 && p.in_nchw)
            idx = (((unsigned)gb * p.IC + ic) * p.IH + ihc) * p.IW + iwc;
          else
            idx = (((unsigned)gb * p.IH + ihc) * p.IW + iwc) * p.IC + ic;
        }
        const float v = p.A[idx];                     // always in-bounds
        const _Float16 hv = (_Float16)(valid ? v : 0.0f);
        if (j < 8) alo[j] = hv; else ahi[j - 8] = hv;
      }
    }
    *(v8h*)&As[pp][arowLocal * 32 + akbase]     = alo;
    *(v8h*)&As[pp][arowLocal * 32 + akbase + 8] = ahi;

    // -------- B fragments straight from swizzled global (guard-free) -------
    // layout: W[ib][n][k%32]; lane fragment = 32 contiguous bytes
    const _Float16* wb  = p.W + (unsigned)ib * wstep;
    const _Float16* b0p = wb + (n0 + lmod) * 32 + (lhi ? 16 : 0);
    const _Float16* b1p = wb + (n0 + 16 + lmod) * 32 + (lhi ? 16 : 0);
    const v8h b00 = *(const v8h*)b0p;
    const v8h b01 = *(const v8h*)(b0p + 8);
    const v8h b10 = *(const v8h*)b1p;
    const v8h b11 = *(const v8h*)(b1p + 8);

    // prefetch next K-block's B fragments (global_prefetch_b8)
    if (ib + 1 < nKb) {
      __builtin_prefetch(b0p + wstep, 0, 1);
      __builtin_prefetch(b1p + wstep, 0, 1);
    }

    __syncthreads();

    // -------- A fragment from LDS (two contiguous 16B reads) --------
    const _Float16* abase = &As[pp][(wave * 16 + lmod) * 32 + (lhi ? 16 : 0)];
    const v8h a0 = *(const v8h*)abase;
    const v8h a1 = *(const v8h*)(abase + 8);
    const v16h af = __builtin_shufflevector(a0, a1, 0, 1, 2, 3, 4, 5, 6, 7,
                                            8, 9, 10, 11, 12, 13, 14, 15);
    const v16h bf0 = __builtin_shufflevector(b00, b01, 0, 1, 2, 3, 4, 5, 6, 7,
                                             8, 9, 10, 11, 12, 13, 14, 15);
    const v16h bf1 = __builtin_shufflevector(b10, b11, 0, 1, 2, 3, 4, 5, 6, 7,
                                             8, 9, 10, 11, 12, 13, 14, 15);

    acc0 = __builtin_amdgcn_wmma_f32_16x16x32_f16(
        false, af, false, bf0, (short)0, acc0, false, false);
    acc1 = __builtin_amdgcn_wmma_f32_16x16x32_f16(
        false, af, false, bf1, (short)0, acc1, false, false);
  }

  // -------- epilogue: hoisted bias + templated activation --------
#pragma unroll
  for (int s = 0; s < 2; ++s) {
    const v8f acc = s ? acc1 : acc0;
    const int n = n0 + s * 16 + lmod;
    const bool nok = n < p.N;
    const float bv = p.bias[min(n, p.N - 1)];   // branchless, hoisted
#pragma unroll
    for (int r = 0; r < 8; ++r) {
      const int m = m0 + wave * 16 + r + (lhi ? 8 : 0);
      if (nok && m < p.M) {
        float v = acc[r] + bv;
        if (ACT == ACT_RELU) v = fmaxf(v, 0.0f);
        else if (ACT == ACT_SOFTPLUS) {
          const float ex = __expf(fminf(v, 20.0f));
          const float sp = log1pf(ex);
          v = ((v > 20.0f) ? v : sp) + 1e-4f;
        }
        p.C[(unsigned)m * (unsigned)p.N + (unsigned)n] = v;
      }
    }
  }
}

// ------------------------- weight packing kernels ---------------------------
// All packers write the swizzled layout idx(k,n) = kb*(Npad*32) + n*32 + k%32
// with zero padding to (Kpad, Npad).

// conv/deconv OIHW f32 -> swizzled [K = 16*IC, N = OC]
__global__ void pack_conv_w_k(const float* w, _Float16* out, int OC, int IC,
                              int Kpad, int Npad) {
  long i = (long)blockIdx.x * blockDim.x + threadIdx.x;
  long tot = (long)Kpad * Npad;
  if (i >= tot) return;
  int kb = (int)(i / ((long)Npad * 32));
  int r  = (int)(i % ((long)Npad * 32));
  int n  = r >> 5;          // oc
  int kk = r & 31;
  int k  = kb * 32 + kk;    // (kh*4+kw)*IC + ic
  float v = 0.0f;
  if (k < IC * 16 && n < OC) {
    int ic = k % IC;
    int q  = k / IC;
    int kh = q >> 2, kw = q & 3;
    v = w[(((long)n * IC + ic) * 4 + kh) * 4 + kw];
  }
  out[i] = (_Float16)v;
}

// [K,N] f32 -> swizzled f16; optional NCHW<->NHWC fold on rows (encoder wf)
// or columns (decoder wf), with C=256, H=W=4.
__global__ void pack_fc_k(const float* w, _Float16* out, int K, int N,
                          int Kpad, int Npad, int rowPerm, int colPerm) {
  long i = (long)blockIdx.x * blockDim.x + threadIdx.x;
  long tot = (long)Kpad * Npad;
  if (i >= tot) return;
  int kb = (int)(i / ((long)Npad * 32));
  int r  = (int)(i % ((long)Npad * 32));
  int n  = r >> 5;
  int kk = r & 31;
  int k  = kb * 32 + kk;
  float v = 0.0f;
  if (k < K && n < N) {
    int ks = k, ns = n;
    if (rowPerm) { int c = k % 256, hw = k / 256; ks = c * 16 + hw; }
    if (colPerm) { int c = n % 256, hw = n / 256; ns = c * 16 + hw; }
    v = w[(long)ks * N + ns];
  }
  out[i] = (_Float16)v;
}

__global__ void perm_bias_k(const float* b, float* out, int N) {
  int i = blockIdx.x * blockDim.x + threadIdx.x;
  if (i >= N) return;
  int c = i % 256, hw = i / 256;
  out[i] = b[c * 16 + hw];
}

// ------------------------- glue / reduction kernels -------------------------

__global__ void concat4_k(float* dst, int Bn, int din,
                          const float* s0, int l0, const float* s1, int l1,
                          const float* s2, int l2, const float* s3, int l3) {
  int i = blockIdx.x * blockDim.x + threadIdx.x;
  if (i >= Bn * din) return;
  int b = i / din, j = i % din;
  float v = 0.0f;
  if (j < l0)               { if (s0) v = s0[b * l0 + j]; }
  else if ((j -= l0) < l1)  { if (s1) v = s1[b * l1 + j]; }
  else if ((j -= l1) < l2)  { if (s2) v = s2[b * l2 + j]; }
  else { j -= l2;             if (s3) v = s3[b * l3 + j]; }
  dst[i] = v;
}

__global__ void rsample_k(const float* loc, const float* sc, const float* eps,
                          float* s, int n) {
  int i = blockIdx.x * blockDim.x + threadIdx.x;
  if (i < n) s[i] = loc[i] + sc[i] * eps[i];
}

__global__ void kl_k(const float* ql, const float* qs, const float* pl,
                     const float* ps, float* loss, int n, float invB) {
  __shared__ float sd[256];
  float acc = 0.0f;
  for (int i = threadIdx.x; i < n; i += blockDim.x) {
    float d  = ql[i] - pl[i];
    float p2 = ps[i] * ps[i];
    acc += __logf(ps[i] / qs[i]) + (qs[i] * qs[i] + d * d) / (2.0f * p2) - 0.5f;
  }
  sd[threadIdx.x] = acc;
  __syncthreads();
  for (int s2 = 128; s2 > 0; s2 >>= 1) {
    if (threadIdx.x < s2) sd[threadIdx.x] += sd[threadIdx.x + s2];
    __syncthreads();
  }
  if (threadIdx.x == 0) atomicAdd(loss, sd[0] * invB);
}

// loss -= mean_b sum_chw [ -0.5 (x - loc)^2 - 0.5 log(2 pi) ]
__global__ void neglogp_k(const float* loc_nhwc, const float* x_nchw,
                          float* loss, float invB) {
  __shared__ float sd[256];
  const int n = 32 * 3 * 64 * 64;
  float acc = 0.0f;
  for (int i = blockIdx.x * blockDim.x + threadIdx.x; i < n;
       i += gridDim.x * blockDim.x) {
    int b = i / 12288;
    int r = i % 12288;
    int c = r / 4096;
    int hw = r % 4096;
    int h = hw >> 6, w = hw & 63;
    float lv = loc_nhwc[(((unsigned)b * 64 + h) * 64 + w) * 3 + c];
    float xv = x_nchw[i];
    float d  = xv - lv;
    acc += -0.5f * d * d - 0.91893853320467274f;
  }
  sd[threadIdx.x] = acc;
  __syncthreads();
  for (int s2 = 128; s2 > 0; s2 >>= 1) {
    if (threadIdx.x < s2) sd[threadIdx.x] += sd[threadIdx.x + s2];
    __syncthreads();
  }
  if (threadIdx.x == 0) atomicAdd(loss, -sd[0] * invB);
}

__global__ void zero_k(float* p) {
  if (threadIdx.x == 0 && blockIdx.x == 0) p[0] = 0.0f;
}

// ---------------------------------------------------------------------------

static inline int ilog2_or_neg1(int v) {
  if (v <= 0 || (v & (v - 1))) return -1;
  int l = 0;
  while ((1 << l) < v) ++l;
  return l;
}
static inline int pad32(int v) { return (v + 31) & ~31; }

extern "C" void kernel_launch(void* const* d_in, const int* in_sizes, int n_in,
                              void* d_out, int out_size, void* d_ws, size_t ws_size,
                              hipStream_t stream) {
  if (n_in < 68) return;

  // Input order: x0, x, a, eps, then params leaves (jax pytree, dict keys
  // sorted): dec0(4..13), dec1(14..23), enc0(24..33), enc1(34..43),
  // post0(44..49), post1(50..55), prior0(56..61), prior1(62..67).
  // Per dec/enc: b1,b2,b3,b4,bf,w1,w2,w3,w4,wf.  Per gauss: b1,bl,bs,w1,wl,ws.
  auto F = [&](int i) { return (const float*)d_in[i]; };
  const float* x0  = F(0);
  const float* x   = F(1);
  const float* a   = F(2);
  const float* eps = F(3);
  float* loss = (float*)d_out;

  const int T = 20, Bn = 32;
  const int din_post[2]  = {64 + 1024 + 4, 64 + 1024 + 4 + 64};
  const int din_prior[2] = {64 + 4,        64 + 4 + 64};

  // ----- workspace bump allocator -----
  char* ws = (char*)d_ws;
  size_t off = 0;
  auto alloc = [&](size_t bytes) -> void* {
    void* p = ws + off;
    off = (off + bytes + 255) & ~(size_t)255;
    return p;
  };
  auto allocH = [&](size_t n) { return (_Float16*)alloc(n * sizeof(_Float16)); };
  auto allocF = [&](size_t n) { return (float*)alloc(n * sizeof(float)); };

  // packed weights (padded swizzled sizes)
  _Float16 *wc1p[2], *wc2p[2], *wc3p[2], *wc4p[2], *wfp[2];
  _Float16 *dwfp[2], *wd1p[2], *wd2p[2], *wd3p[2], *wd4p[2];
  float* dbfp[2];
  for (int e = 0; e < 2; ++e) {
    wc1p[e] = allocH((size_t)pad32(48) * 32);
    wc2p[e] = allocH((size_t)512 * 64);
    wc3p[e] = allocH((size_t)1024 * 128);
    wc4p[e] = allocH((size_t)2048 * 256);
    wfp[e]  = allocH((size_t)4096 * 1024);
    dwfp[e] = allocH((size_t)64 * 4096);
    dbfp[e] = allocF(4096);
    wd1p[e] = allocH((size_t)4096 * 128);
    wd2p[e] = allocH((size_t)2048 * 64);
    wd3p[e] = allocH((size_t)1024 * 32);
    wd4p[e] = allocH((size_t)512 * 32);
  }
  _Float16 *pw1[2], *pwl[2], *pws[2], *rw1[2], *rwl[2], *rws[2];
  for (int i = 0; i < 2; ++i) {
    pw1[i] = allocH((size_t)pad32(din_post[i]) * 256);
    pwl[i] = allocH((size_t)256 * 64);
    pws[i] = allocH((size_t)256 * 64);
    rw1[i] = allocH((size_t)pad32(din_prior[i]) * 256);
    rwl[i] = allocH((size_t)256 * 64);
    rws[i] = allocH((size_t)256 * 64);
  }
  // activations (f32)
  float* h1   = allocF((size_t)Bn * 32 * 32 * 32);
  float* h2   = allocF((size_t)Bn * 16 * 16 * 64);
  float* h3   = allocF((size_t)Bn * 8 * 8 * 128);
  float* h4   = allocF((size_t)Bn * 4 * 4 * 256);
  float* henc = allocF((size_t)Bn * 1024);
  float* h0   = allocF((size_t)Bn * 4 * 4 * 256);
  float* d1b  = allocF((size_t)Bn * 8 * 8 * 128);
  float* d2b  = allocF((size_t)Bn * 16 * 16 * 64);
  float* d3b  = allocF((size_t)Bn * 32 * 32 * 32);
  float* locb = allocF((size_t)Bn * 64 * 64 * 3);
  float* catb = allocF((size_t)Bn * 1156);
  float* gh   = allocF((size_t)Bn * 256);
  float* qloc = allocF(Bn * 64);
  float* qsc  = allocF(Bn * 64);
  float* ploc = allocF(Bn * 64);
  float* psc  = allocF(Bn * 64);
  float* sbuf[4];
  for (int i = 0; i < 4; ++i) sbuf[i] = allocF(Bn * 64);
  if (off > ws_size) return;  // workspace too small: bail before any write

  auto launch1d = [&](auto kern, long n, auto... args) {
    int blocks = (int)((n + 255) / 256);
    hipLaunchKernelGGL(kern, dim3(blocks), dim3(256), 0, stream, args...);
  };
  auto packConv = [&](const float* w, _Float16* dst, int OC, int IC) {
    int Kp = pad32(IC * 16), Np = pad32(OC);
    launch1d(pack_conv_w_k, (long)Kp * Np, w, dst, OC, IC, Kp, Np);
  };
  auto packFc = [&](const float* w, _Float16* dst, int K, int N,
                    int rowPerm, int colPerm) {
    int Kp = pad32(K), Np = pad32(N);
    launch1d(pack_fc_k, (long)Kp * Np, w, dst, K, N, Kp, Np, rowPerm, colPerm);
  };
  auto gemm = [&](int mode, int act, const float* A, const _Float16* W,
                  const float* bias, float* C, int M, int N, int K,
                  int IC, int IH, int OHW, int in_nchw) {
    GemmP p{A, W, bias, C, M, N, K, pad32(N),
            IC, ilog2_or_neg1(IC), IH, IH, OHW, in_nchw};
    dim3 g(((M + 63) / 64) * (pad32(N) / 32)), b(128);
    if (mode == 0) {
      if (act == ACT_NONE)
        hipLaunchKernelGGL((gemm_wmma_k<0, ACT_NONE>), g, b, 0, stream, p);
      else if (act == ACT_RELU)
        hipLaunchKernelGGL((gemm_wmma_k<0, ACT_RELU>), g, b, 0, stream, p);
      else
        hipLaunchKernelGGL((gemm_wmma_k<0, ACT_SOFTPLUS>), g, b, 0, stream, p);
    } else if (mode == 1) {
      hipLaunchKernelGGL((gemm_wmma_k<1, ACT_RELU>), g, b, 0, stream, p);
    } else {
      if (act == ACT_RELU)
        hipLaunchKernelGGL((gemm_wmma_k<2, ACT_RELU>), g, b, 0, stream, p);
      else
        hipLaunchKernelGGL((gemm_wmma_k<2, ACT_NONE>), g, b, 0, stream, p);
    }
  };

  hipLaunchKernelGGL(zero_k, dim3(1), dim3(32), 0, stream, loss);

  // ----- pack weights (every call: no cached state) -----
  for (int e = 0; e < 2; ++e) {
    int eb = 24 + e * 10;  // enc leaves
    packConv(F(eb + 5), wc1p[e], 32, 3);
    packConv(F(eb + 6), wc2p[e], 64, 32);
    packConv(F(eb + 7), wc3p[e], 128, 64);
    packConv(F(eb + 8), wc4p[e], 256, 128);
    packFc(F(eb + 9), wfp[e], 4096, 1024, 1, 0);
    int db = 4 + e * 10;   // dec leaves
    packFc(F(db + 9), dwfp[e], 64, 4096, 0, 1);
    launch1d(perm_bias_k, 4096, F(db + 4), dbfp[e], 4096);
    packConv(F(db + 5), wd1p[e], 128, 256);
    packConv(F(db + 6), wd2p[e], 64, 128);
    packConv(F(db + 7), wd3p[e], 32, 64);
    packConv(F(db + 8), wd4p[e], 3, 32);
  }
  for (int i = 0; i < 2; ++i) {
    int pb = 44 + i * 6, rb = 56 + i * 6;
    packFc(F(pb + 3), pw1[i], din_post[i], 256, 0, 0);
    packFc(F(pb + 4), pwl[i], 256, 64, 0, 0);
    packFc(F(pb + 5), pws[i], 256, 64, 0, 0);
    packFc(F(rb + 3), rw1[i], din_prior[i], 256, 0, 0);
    packFc(F(rb + 4), rwl[i], 256, 64, 0, 0);
    packFc(F(rb + 5), rws[i], 256, 64, 0, 0);
  }

  auto encode = [&](int e, const float* xin) {
    int eb = 24 + e * 10;
    gemm(1, ACT_RELU, xin, wc1p[e], F(eb + 0), h1, 32768, 32, 48,  3,   64, 32, 1);
    gemm(1, ACT_RELU, h1,  wc2p[e], F(eb + 1), h2, 8192, 64, 512,  32,  32, 16, 0);
    gemm(1, ACT_RELU, h2,  wc3p[e], F(eb + 2), h3, 2048, 128, 1024, 64, 16, 8, 0);
    gemm(1, ACT_RELU, h3,  wc4p[e], F(eb + 3), h4, 512, 256, 2048, 128, 8, 4, 0);
    gemm(0, ACT_NONE, h4,  wfp[e],  F(eb + 4), henc, 32, 1024, 4096, 0, 0, 0, 0);
  };
  auto gauss = [&](const float* cat, int din, int base, const _Float16* w1,
                   const _Float16* wl, const _Float16* wsp_, float* locO, float* scO) {
    gemm(0, ACT_RELU, cat, w1, F(base + 0), gh, 32, 256, din, 0, 0, 0, 0);
    gemm(0, ACT_NONE, gh, wl, F(base + 1), locO, 32, 64, 256, 0, 0, 0, 0);
    if (scO) gemm(0, ACT_SOFTPLUS, gh, wsp_, F(base + 2), scO, 32, 64, 256, 0, 0, 0, 0);
  };
  auto decode = [&](int e, const float* s) {
    int db = 4 + e * 10;
    gemm(0, ACT_RELU, s,   dwfp[e], dbfp[e],  h0,  32, 4096, 64,    0, 0, 0, 0);
    gemm(2, ACT_RELU, h0,  wd1p[e], F(db + 0), d1b, 2048, 128, 4096, 256, 4, 8, 0);
    gemm(2, ACT_RELU, d1b, wd2p[e], F(db + 1), d2b, 8192, 64, 2048,  128, 8, 16, 0);
    gemm(2, ACT_RELU, d2b, wd3p[e], F(db + 2), d3b, 32768, 32, 1024, 64, 16, 32, 0);
    gemm(2, ACT_NONE, d3b, wd4p[e], F(db + 3), locb, 131072, 3, 512, 32, 32, 64, 0);
  };

  // ----- sample_s0: posterior means with zero s_prev / zero action -----
  float* sp[2] = {sbuf[0], sbuf[1]};
  float* st[2] = {sbuf[2], sbuf[3]};
  for (int i = 0; i < 2; ++i) {
    encode(i, x0);
    int din = din_post[i];
    launch1d(concat4_k, (long)Bn * din, catb, Bn, din,
             (const float*)nullptr, 64, (const float*)henc, 1024,
             (const float*)nullptr, 4 + (i ? 64 : 0), (const float*)nullptr, 0);
    gauss(catb, din, 44 + i * 6, pw1[i], pwl[i], pws[i], sp[i], (float*)nullptr);
  }

  // ----- scan over time -----
  const float invB = 1.0f / Bn;
  for (int t = 0; t < T; ++t) {
    const float* xt = x + (size_t)t * Bn * 3 * 64 * 64;
    const float* at = a + (size_t)t * Bn * 4;
    for (int i = 0; i < 2; ++i) {
      encode(i, xt);
      // posterior: concat [s_prev_i, h, a_t, (s_t0)]
      int dp = din_post[i];
      launch1d(concat4_k, (long)Bn * dp, catb, Bn, dp,
               (const float*)sp[i], 64, (const float*)henc, 1024,
               at, 4, (const float*)(i ? st[0] : nullptr), i ? 64 : 0);
      gauss(catb, dp, 44 + i * 6, pw1[i], pwl[i], pws[i], qloc, qsc);
      // prior: concat [s_prev_i, a_t, (s_t0)]
      int dr = din_prior[i];
      launch1d(concat4_k, (long)Bn * dr, catb, Bn, dr,
               (const float*)sp[i], 64, at, 4,
               (const float*)(i ? st[0] : nullptr), i ? 64 : 0,
               (const float*)nullptr, 0);
      gauss(catb, dr, 56 + i * 6, rw1[i], rwl[i], rws[i], ploc, psc);
      // KL(q || p)
      hipLaunchKernelGGL(kl_k, dim3(1), dim3(256), 0, stream,
                         (const float*)qloc, (const float*)qsc,
                         (const float*)ploc, (const float*)psc, loss,
                         Bn * 64, invB);
      // rsample
      const float* et = eps + ((size_t)(t * 2 + i) * Bn) * 64;
      launch1d(rsample_k, Bn * 64, (const float*)qloc, (const float*)qsc, et,
               st[i], Bn * 64);
      // reconstruction terms
      decode(i, st[i]);
      hipLaunchKernelGGL(neglogp_k, dim3(256), dim3(256), 0, stream,
                         (const float*)locb, xt, loss, invB);
      decode(i, ploc);
      hipLaunchKernelGGL(neglogp_k, dim3(256), dim3(256), 0, stream,
                         (const float*)locb, xt, loss, invB);
    }
    std::swap(sp[0], st[0]);
    std::swap(sp[1], st[1]);
  }
  (void)in_sizes; (void)out_size;
}
